// MultiHeadAttention_20478404067848
// MI455X (gfx1250) — matrix operand order repaired and checked by reading, not compiled, not run on hardware
//
#include <hip/hip_runtime.h>
#include <stdint.h>

#define SLEN   2048
#define DMODEL 768
#define NHEAD  12
#define DHEAD  64
#define BATCH  4

typedef __bf16 bf16;
typedef __attribute__((ext_vector_type(16))) __bf16 v16bf;
typedef __attribute__((ext_vector_type(8)))  float  v8f;

// ---------------------------------------------------------------------------
// CDNA5 async global->LDS copy (no VGPR staging; tracked by ASYNCcnt)
// ---------------------------------------------------------------------------
static __device__ inline void async_copy_b128(void* lds_dst, const void* gsrc) {
  // low 32 bits of a flat-cast __shared__ pointer == workgroup-relative LDS
  // byte offset, which HW adds to the wave's LDS_BASE.
  unsigned loff = (unsigned)(uintptr_t)lds_dst;
  asm volatile("global_load_async_to_lds_b128 %0, %1, off"
               :: "v"(loff), "v"(gsrc) : "memory");
}
static __device__ inline void async_wait0() {
  asm volatile("s_wait_asynccnt 0" ::: "memory");
}

// ---------------------------------------------------------------------------
// WMMA helpers (CDNA5 layouts, wave32)
// ---------------------------------------------------------------------------
static __device__ inline v8f wmma_bf16(v16bf a, v16bf b, v8f c) {
  // D = A(16x32 bf16) * B(32x16 bf16) + C(16x16 f32)
  return __builtin_amdgcn_wmma_f32_16x16x32_bf16(
      false, a, false, b, (short)0, c, false, false);
}

// A-matrix 16x32 bf16 fragment. Per ISA: lanes 0-15 -> M, VGPR v holds K pair
// (v<4 ? 2v : 16+2(v-4)) + 8*(lane/16).
static __device__ inline v16bf load_a_frag(const bf16* base, int rowStride,
                                           int lane) {
  const int m = lane & 15, g = lane >> 4;
  const bf16* ap = base + m * rowStride;
  v16bf af;
#pragma unroll
  for (int v = 0; v < 8; ++v) {
    int kk = ((v < 4) ? 2 * v : 16 + 2 * (v - 4)) + 8 * g;
    af[2 * v]     = ap[kk];
    af[2 * v + 1] = ap[kk + 1];
  }
  return af;
}

// B-matrix 32x16 bf16 fragment: lanes 0-15 -> N, element e holds K
// (e<8 ? e : 16+(e-8)) + 8*(lane/16), matching the A fragment.
static __device__ inline v16bf load_b_frag(const bf16* base, int rowStride,
                                           int lane) {
  const int n = lane & 15, g = lane >> 4;
  v16bf bfr;
#pragma unroll
  for (int e = 0; e < 16; ++e) {
    int kk = ((e < 8) ? e : 16 + (e - 8)) + 8 * g;
    bfr[e] = base[kk * rowStride + n];
  }
  return bfr;
}

// ---------------------------------------------------------------------------
// fp32 -> bf16 cast
// ---------------------------------------------------------------------------
__global__ __launch_bounds__(256) void cast_f32_bf16(
    const float* __restrict__ s, bf16* __restrict__ d, long n) {
  long i = ((long)blockIdx.x * blockDim.x + threadIdx.x) * 4;
  if (i + 3 < n) {
    float4 f = *(const float4*)(s + i);
    d[i + 0] = (bf16)f.x; d[i + 1] = (bf16)f.y;
    d[i + 2] = (bf16)f.z; d[i + 3] = (bf16)f.w;
  } else {
    for (long j = i; j < n; ++j) d[j] = (bf16)s[j];
  }
}

// ---------------------------------------------------------------------------
// GEMM: [M,768] bf16 x [768,768] bf16 + bias(f32).
// HEADSPLIT=1: write bf16 to [B,H,S,DH]; else write fp32 to [M,768].
// Block = 8 waves -> 128x64 tile; K slices streamed to LDS by the async unit.
// ---------------------------------------------------------------------------
template <int HEADSPLIT>
__global__ __launch_bounds__(256) void gemm768(
    const bf16* __restrict__ A, const bf16* __restrict__ W,
    const float* __restrict__ bias, bf16* __restrict__ outBf,
    float* __restrict__ outF, int M) {
  __shared__ __align__(16) bf16 lA[128 * 32];
  __shared__ __align__(16) bf16 lB[32 * 64];
  const int tid = threadIdx.x, wave = tid >> 5, lane = tid & 31;
  const int m0 = blockIdx.x * 128, n0 = blockIdx.y * 64;
  const int nl = lane & 15, g = lane >> 4;

  v8f acc[4] = {};
  for (int k0 = 0; k0 < DMODEL; k0 += 32) {
    __syncthreads();
    {  // A tile 128x32: 256 threads x 16 bf16, direct to LDS
      int r = tid >> 1, half = tid & 1;
      const bf16* gp = A + (long)(m0 + r) * DMODEL + k0 + half * 16;
      bf16* lp = lA + r * 32 + half * 16;
      async_copy_b128(lp, gp);
      async_copy_b128(lp + 8, gp + 8);
      if (k0 + 32 < DMODEL)  // pull next slice toward L2
        __builtin_prefetch(gp + 32, 0, 1);
    }
    if (tid < 128) {  // B tile 32x64
      int r = tid >> 2, c = (tid & 3) * 16;
      const bf16* gp = W + (long)(k0 + r) * DMODEL + n0 + c;
      bf16* lp = lB + r * 64 + c;
      async_copy_b128(lp, gp);
      async_copy_b128(lp + 8, gp + 8);
    }
    async_wait0();
    __syncthreads();

    v16bf af = load_a_frag(lA + wave * 16 * 32, 32, lane);
#pragma unroll
    for (int ct = 0; ct < 4; ++ct) {
      v16bf bfr = load_b_frag(lB + ct * 16, 64, lane);
      acc[ct] = wmma_bf16(af, bfr, acc[ct]);
    }
  }

#pragma unroll
  for (int ct = 0; ct < 4; ++ct) {
    int n = n0 + ct * 16 + nl;
    float bv = bias[n];
#pragma unroll
    for (int r = 0; r < 8; ++r) {
      int m = m0 + wave * 16 + r + 8 * g;
      float val = acc[ct][r] + bv;
      if (HEADSPLIT) {
        int bb = m / SLEN, s = m % SLEN;
        int h = n / DHEAD, dh = n % DHEAD;
        outBf[(((long)bb * NHEAD + h) * SLEN + s) * DHEAD + dh] = (bf16)val;
      } else {
        outF[(long)m * DMODEL + n] = val;
      }
    }
  }
  (void)M;
}

// ---------------------------------------------------------------------------
// Pass 1: flash attention — per (b,h, 64-row q block): online softmax stats
// (m,l) + context P@V accumulated in registers. 4 waves, 16 rows per wave.
// V/Q tiles via async DMA; K transposed through VGPRs (overlaps the async V).
// ---------------------------------------------------------------------------
__global__ __launch_bounds__(128) void flash_fwd(
    const bf16* __restrict__ Qh, const bf16* __restrict__ Kh,
    const bf16* __restrict__ Vh, bf16* __restrict__ ctx,
    float* __restrict__ rowM, float* __restrict__ rowL) {
  __shared__ __align__(16) bf16 lQ[64 * DHEAD];
  __shared__ __align__(16) bf16 lKt[DHEAD * 128];  // [dh][col]
  __shared__ __align__(16) bf16 lV[128 * DHEAD];   // [col][dh]
  __shared__ __align__(16) bf16 lP[4 * 16 * 128];  // per-wave P buffer

  const int bh = blockIdx.x;
  const int q0 = blockIdx.y * 64;
  const int tid = threadIdx.x, wave = tid >> 5, lane = tid & 31;
  const int nl = lane & 15, g = lane >> 4;

  const bf16* Qp = Qh + (long)bh * SLEN * DHEAD;
  const bf16* Kp = Kh + (long)bh * SLEN * DHEAD;
  const bf16* Vp = Vh + (long)bh * SLEN * DHEAD;

  {  // Q tile 64x64: async, 2 threads/row, 32 bf16 each
    int r = tid >> 1, half = tid & 1;
    const bf16* gp = Qp + (long)(q0 + r) * DHEAD + half * 32;
    bf16* lp = lQ + r * DHEAD + half * 32;
#pragma unroll
    for (int i = 0; i < 4; ++i) async_copy_b128(lp + i * 8, gp + i * 8);
  }
  async_wait0();
  __syncthreads();

  const v16bf aq0 = load_a_frag(lQ + wave * 16 * DHEAD, DHEAD, lane);
  const v16bf aq1 = load_a_frag(lQ + wave * 16 * DHEAD + 32, DHEAD, lane);

  float m_i[8], l_i[8];
#pragma unroll
  for (int r = 0; r < 8; ++r) { m_i[r] = -3.0e38f; l_i[r] = 0.f; }
  v8f cacc[4] = {};
  bf16* lPw = lP + wave * 16 * 128;

  for (int j = 0; j < q0 + 64; j += 128) {  // causal column range
    __syncthreads();
    {  // V tile: async DMA runs while we transpose K below
      int c = tid;
      const bf16* gp = Vp + (long)(j + c) * DHEAD;
      bf16* lp = lV + c * DHEAD;
#pragma unroll
      for (int i = 0; i < 8; ++i) async_copy_b128(lp + i * 8, gp + i * 8);
    }
    {  // K^T tile: transpose on the way into LDS
      int c = tid;
      const bf16* src = Kp + (long)(j + c) * DHEAD;
#pragma unroll
      for (int dh0 = 0; dh0 < DHEAD; dh0 += 8) {
        uint4 dta = *(const uint4*)(src + dh0);
        const bf16* e = (const bf16*)&dta;
#pragma unroll
        for (int i = 0; i < 8; ++i) lKt[(dh0 + i) * 128 + c] = e[i];
      }
    }
    async_wait0();
    __syncthreads();

    // logits 16x128: 8 col tiles x 2 WMMA (K=64 split in two x32)
    v8f sc[8];
#pragma unroll
    for (int ct = 0; ct < 8; ++ct) {
      v8f s = {};
      s = wmma_bf16(aq0, load_b_frag(lKt + ct * 16, 128, lane), s);
      s = wmma_bf16(aq1, load_b_frag(lKt + 32 * 128 + ct * 16, 128, lane), s);
      sc[ct] = s;
    }

    // scale + causal mask
#pragma unroll
    for (int ct = 0; ct < 8; ++ct) {
      int col = j + ct * 16 + nl;
#pragma unroll
      for (int r = 0; r < 8; ++r) {
        int row = q0 + wave * 16 + r + 8 * g;
        float s = sc[ct][r] * 0.125f;  // 1/sqrt(64)
        sc[ct][r] = (col > row) ? -3.0e38f : s;
      }
    }

    // online softmax update (row groups span 16 lanes -> shfl_xor reduce)
#pragma unroll
    for (int r = 0; r < 8; ++r) {
      float tmax = -3.0e38f;
#pragma unroll
      for (int ct = 0; ct < 8; ++ct) tmax = fmaxf(tmax, sc[ct][r]);
      tmax = fmaxf(tmax, __shfl_xor(tmax, 1, 32));
      tmax = fmaxf(tmax, __shfl_xor(tmax, 2, 32));
      tmax = fmaxf(tmax, __shfl_xor(tmax, 4, 32));
      tmax = fmaxf(tmax, __shfl_xor(tmax, 8, 32));
      float mnew = fmaxf(m_i[r], tmax);
      float corr = __expf(m_i[r] - mnew);
      float psum = 0.f;
#pragma unroll
      for (int ct = 0; ct < 8; ++ct) {
        float p = __expf(sc[ct][r] - mnew);
        sc[ct][r] = p;
        psum += p;
      }
      psum += __shfl_xor(psum, 1, 32);
      psum += __shfl_xor(psum, 2, 32);
      psum += __shfl_xor(psum, 4, 32);
      psum += __shfl_xor(psum, 8, 32);
      l_i[r] = l_i[r] * corr + psum;
      m_i[r] = mnew;
#pragma unroll
      for (int ctd = 0; ctd < 4; ++ctd) cacc[ctd][r] *= corr;
    }

    // P (C layout) -> LDS -> A layout for ctx += P @ V
#pragma unroll
    for (int ct = 0; ct < 8; ++ct)
#pragma unroll
      for (int r = 0; r < 8; ++r)
        lPw[(r + 8 * g) * 128 + ct * 16 + nl] = (bf16)sc[ct][r];

#pragma unroll
    for (int chunk = 0; chunk < 4; ++chunk) {
      v16bf pa = load_a_frag(lPw + chunk * 32, 128, lane);
#pragma unroll
      for (int ctd = 0; ctd < 4; ++ctd) {
        v16bf vb = load_b_frag(lV + chunk * 32 * DHEAD + ctd * 16, DHEAD, lane);
        cacc[ctd] = wmma_bf16(pa, vb, cacc[ctd]);
      }
    }
  }

  // finalize: ctx (merged-head [B,S,768] bf16) + stats
  const int b = bh / NHEAD, h = bh % NHEAD;
#pragma unroll
  for (int r = 0; r < 8; ++r) {
    int row = q0 + wave * 16 + r + 8 * g;
    float inv = 1.0f / l_i[r];
#pragma unroll
    for (int ctd = 0; ctd < 4; ++ctd) {
      float val = cacc[ctd][r] * inv;
      ctx[((long)b * SLEN + row) * DMODEL + h * DHEAD + ctd * 16 + nl] =
          (bf16)val;
    }
    if (nl == 0) {
      rowM[(long)bh * SLEN + row] = m_i[r];
      rowL[(long)bh * SLEN + row] = l_i[r];
    }
  }
}

// ---------------------------------------------------------------------------
// Pass 2: recompute logits, normalize with saved stats, stream attn (fp32).
// Full column range so masked entries are written as exact zeros.
// ---------------------------------------------------------------------------
__global__ __launch_bounds__(128) void attn_write(
    const bf16* __restrict__ Qh, const bf16* __restrict__ Kh,
    const float* __restrict__ rowM, const float* __restrict__ rowL,
    float* __restrict__ attn) {
  __shared__ __align__(16) bf16 lQ[64 * DHEAD];
  __shared__ __align__(16) bf16 lKt[DHEAD * 128];

  const int bh = blockIdx.x;
  const int q0 = blockIdx.y * 64;
  const int tid = threadIdx.x, wave = tid >> 5, lane = tid & 31;
  const int nl = lane & 15, g = lane >> 4;

  const bf16* Qp = Qh + (long)bh * SLEN * DHEAD;
  const bf16* Kp = Kh + (long)bh * SLEN * DHEAD;

  {  // Q tile async
    int r = tid >> 1, half = tid & 1;
    const bf16* gp = Qp + (long)(q0 + r) * DHEAD + half * 32;
    bf16* lp = lQ + r * DHEAD + half * 32;
#pragma unroll
    for (int i = 0; i < 4; ++i) async_copy_b128(lp + i * 8, gp + i * 8);
  }
  async_wait0();
  __syncthreads();

  const v16bf aq0 = load_a_frag(lQ + wave * 16 * DHEAD, DHEAD, lane);
  const v16bf aq1 = load_a_frag(lQ + wave * 16 * DHEAD + 32, DHEAD, lane);

  float m_i[8], linv[8];
#pragma unroll
  for (int r = 0; r < 8; ++r) {
    int row = q0 + wave * 16 + r + 8 * g;
    m_i[r] = rowM[(long)bh * SLEN + row];
    linv[r] = 1.0f / rowL[(long)bh * SLEN + row];
  }

  for (int j = 0; j < SLEN; j += 128) {
    __syncthreads();
    {
      int c = tid;
      const bf16* src = Kp + (long)(j + c) * DHEAD;
#pragma unroll
      for (int dh0 = 0; dh0 < DHEAD; dh0 += 8) {
        uint4 dta = *(const uint4*)(src + dh0);
        const bf16* e = (const bf16*)&dta;
#pragma unroll
        for (int i = 0; i < 8; ++i) lKt[(dh0 + i) * 128 + c] = e[i];
      }
    }
    __syncthreads();

#pragma unroll
    for (int ct = 0; ct < 8; ++ct) {
      v8f s = {};
      s = wmma_bf16(aq0, load_b_frag(lKt + ct * 16, 128, lane), s);
      s = wmma_bf16(aq1, load_b_frag(lKt + 32 * 128 + ct * 16, 128, lane), s);
      int col = j + ct * 16 + nl;
#pragma unroll
      for (int r = 0; r < 8; ++r) {
        int row = q0 + wave * 16 + r + 8 * g;
        float p = (col > row) ? 0.0f
                              : __expf(s[r] * 0.125f - m_i[r]) * linv[r];
        attn[((long)bh * SLEN + row) * SLEN + col] = p;
      }
    }
  }
}

// ---------------------------------------------------------------------------
// Host-side orchestration
// ---------------------------------------------------------------------------
extern "C" void kernel_launch(void* const* d_in, const int* in_sizes, int n_in,
                              void* d_out, int out_size, void* d_ws,
                              size_t ws_size, hipStream_t stream) {
  (void)in_sizes; (void)n_in; (void)out_size; (void)ws_size;
  const float* v    = (const float*)d_in[0];
  const float* k    = (const float*)d_in[1];
  const float* q    = (const float*)d_in[2];
  /* d_in[3] = mask (causal, handled analytically) */
  const float* wq_w = (const float*)d_in[4];
  const float* wq_b = (const float*)d_in[5];
  const float* wk_w = (const float*)d_in[6];
  const float* wk_b = (const float*)d_in[7];
  const float* wv_w = (const float*)d_in[8];
  const float* wv_b = (const float*)d_in[9];
  const float* wo_w = (const float*)d_in[10];
  const float* wo_b = (const float*)d_in[11];

  const long ND = (long)BATCH * SLEN * DMODEL;   // 6,291,456
  const long WN = (long)DMODEL * DMODEL;         //   589,824
  const long SN = (long)BATCH * NHEAD * SLEN;    //    98,304

  char* w = (char*)d_ws;
  bf16* qb  = (bf16*)w; w += ND * 2;
  bf16* kb  = (bf16*)w; w += ND * 2;
  bf16* vb  = (bf16*)w; w += ND * 2;
  bf16* wqb = (bf16*)w; w += WN * 2;
  bf16* wkb = (bf16*)w; w += WN * 2;
  bf16* wvb = (bf16*)w; w += WN * 2;
  bf16* wob = (bf16*)w; w += WN * 2;
  bf16* Qh  = (bf16*)w; w += ND * 2;
  bf16* Kh  = (bf16*)w; w += ND * 2;
  bf16* Vh  = (bf16*)w; w += ND * 2;
  bf16* ctx = (bf16*)w; w += ND * 2;
  float* rowM = (float*)w; w += SN * 4;
  float* rowL = (float*)w; w += SN * 4;

  float* outO = (float*)d_out;       // [B,S,D]
  float* outA = outO + ND;           // [B,H,S,S]

  auto cast = [&](const float* s, bf16* d, long n) {
    long v4 = (n + 3) / 4;
    unsigned gsz = (unsigned)((v4 + 255) / 256);
    cast_f32_bf16<<<dim3(gsz), dim3(256), 0, stream>>>(s, d, n);
  };
  cast(q, qb, ND);    cast(k, kb, ND);    cast(v, vb, ND);
  cast(wq_w, wqb, WN); cast(wk_w, wkb, WN);
  cast(wv_w, wvb, WN); cast(wo_w, wob, WN);

  dim3 ggrid(64, 12);  // 8192/128 x 768/64
  gemm768<1><<<ggrid, 256, 0, stream>>>(qb, wqb, wq_b, Qh, nullptr,
                                        BATCH * SLEN);
  gemm768<1><<<ggrid, 256, 0, stream>>>(kb, wkb, wk_b, Kh, nullptr,
                                        BATCH * SLEN);
  gemm768<1><<<ggrid, 256, 0, stream>>>(vb, wvb, wv_b, Vh, nullptr,
                                        BATCH * SLEN);

  dim3 agrid(BATCH * NHEAD, SLEN / 64);  // 48 x 32
  flash_fwd<<<agrid, 128, 0, stream>>>(Qh, Kh, Vh, ctx, rowM, rowL);
  attn_write<<<agrid, 128, 0, stream>>>(Qh, Kh, rowM, rowL, outA);

  gemm768<0><<<ggrid, 256, 0, stream>>>(ctx, wob, wo_b, nullptr, outO,
                                        BATCH * SLEN);
}
